// FeatureExtraction_64424509440083
// MI455X (gfx1250) — matrix operand order repaired and checked by reading, not compiled, not run on hardware
//
#include <hip/hip_runtime.h>

typedef __attribute__((ext_vector_type(16))) _Float16 v16h;
typedef __attribute__((ext_vector_type(8)))  float    v8f;

#define IMG_N 1024
#define IMG_PIX (IMG_N * IMG_N)
#define NBATCH 16

// ---- gfx1250 async global->LDS copy path (ASYNCcnt), guarded ----
#if defined(__gfx1250__) && \
    __has_builtin(__builtin_amdgcn_global_load_async_to_lds_b32) && \
    __has_builtin(__builtin_amdgcn_s_wait_asynccnt)
#define USE_ASYNC_LDS 1
#else
#define USE_ASYNC_LDS 0
#endif

#if USE_ASYNC_LDS
typedef __attribute__((address_space(1))) int GlobI;   // clang prints AS1 as __device__
typedef __attribute__((address_space(3))) int LdsI;    // clang prints AS3 as __shared__
// Global flat addr == AS1 value; LDS flat addr low 32 bits == AS3 offset
// (ISA 10.2: LDS aperture puts the LDS byte address in addr[31:0]).
__device__ __forceinline__ void async_copy_b32(const float* gsrc, float* ldst) {
  __builtin_amdgcn_global_load_async_to_lds_b32(
      (GlobI*)(uintptr_t)gsrc, (LdsI*)(unsigned)(uintptr_t)ldst, 0, 0);
}
#endif

// ---- order-preserving float <-> uint mapping for atomic max ----
__device__ __forceinline__ unsigned enc_mono(float f) {
  unsigned u = __float_as_uint(f);
  return (u & 0x80000000u) ? ~u : (u | 0x80000000u);
}
__device__ __forceinline__ float dec_mono(unsigned e) {
  unsigned u = (e & 0x80000000u) ? (e & 0x7FFFFFFFu) : ~e;
  return __uint_as_float(u);
}

// 16-bit A-matrix (16x32) element -> K index, per CDNA5 ISA layout table.
// element e in v16h maps to VGPR v=e/2, half=e&1; lanes split h = lane>>4.
__device__ __forceinline__ int ka_index(int e, int h) {
  int v = e >> 1;
  int base = (v < 4) ? (v << 1) : (16 + ((v - 4) << 1));
  return base + (h << 3) + (e & 1);
}

// -------------------- K0: init max slots --------------------
__global__ void k_init(unsigned* __restrict__ slots) {
  if (threadIdx.x < 32) slots[threadIdx.x] = 0u;  // dec_mono(0) sorts below everything
}

// -------------------- K1: per-batch input max --------------------
__global__ __launch_bounds__(256) void k_imgmax(const float* __restrict__ img,
                                                unsigned* __restrict__ slots) {
  int b = blockIdx.y;
  const float* p = img + (size_t)b * IMG_PIX;
  float m = -3.4e38f;
  for (int i = blockIdx.x * blockDim.x + threadIdx.x; i < IMG_PIX;
       i += gridDim.x * blockDim.x)
    m = fmaxf(m, p[i]);
  __shared__ float red[256];
  red[threadIdx.x] = m;
  __syncthreads();
  for (int s = 128; s > 0; s >>= 1) {
    if (threadIdx.x < s) red[threadIdx.x] = fmaxf(red[threadIdx.x], red[threadIdx.x + s]);
    __syncthreads();
  }
  if (threadIdx.x == 0) atomicMax(&slots[b], enc_mono(red[0]));
}

// -------------------- K2: WMMA Laplacian-of-Gaussian pass --------------------
// Laplacian_ks(x) = (boxsum_ks(x) - ks^2 * x) / (2*(ks^2-1)), box sums via
// banded matmuls on the WMMA pipe. Block = 32x32 tile, 4 waves x one 16x16.
__global__ __launch_bounds__(128) void k_log(const float* __restrict__ img,
                                             const unsigned* __restrict__ imgmax,
                                             unsigned* __restrict__ logmax,
                                             float* __restrict__ logArr) {
  int b = blockIdx.z;
  int R0 = blockIdx.y * 32, C0 = blockIdx.x * 32;
  float thrImg = dec_mono(imgmax[b]) * 0.05f;

  __shared__ _Float16 inpatch[48 * 48];        // rows/cols [R0-8, R0+39]
  __shared__ _Float16 cs[4][4][16 * 32];       // [wave][size][16 rows x 32 cols]
  __shared__ float red[128];

  const float* p = img + (size_t)b * IMG_PIX;
  for (int idx = threadIdx.x; idx < 48 * 48; idx += 128) {
    int i = idx / 48, j = idx % 48;
    int gr = R0 - 8 + i, gc = C0 - 8 + j;
    float v = 0.f;  // zero padding (border_type='constant')
    if (gr >= 0 && gr < IMG_N && gc >= 0 && gc < IMG_N) {
      float t = p[(size_t)gr * IMG_N + gc];
      v = (t > thrImg) ? t : 0.f;              // per-batch input threshold
    }
    inpatch[idx] = (_Float16)v;
  }
  __syncthreads();

  int w = threadIdx.x >> 5;
  int lane = threadIdx.x & 31;
  int h = lane >> 4;           // lane half
  int n = lane & 15;           // M for A-operand, N for B/C/D operands
  int tr = (w >> 1) * 16, tc = (w & 1) * 16;   // tile origin inside block tile

  // ---- column pass: cs[r][m][c] = sum_{rows m-r..m+r} x, for 32 cols ----
#pragma unroll
  for (int ridx = 0; ridx < 4; ++ridx) {
    int r = ridx + 1;
    v16h a;                                    // band matrix (constant per lane)
#pragma unroll
    for (int e = 0; e < 16; ++e) {
      int K = ka_index(e, h);
      a[e] = (K >= n + 8 - r && K <= n + 8 + r) ? (_Float16)1.f : (_Float16)0.f;
    }
#pragma unroll
    for (int hc = 0; hc < 2; ++hc) {
      v16h bv;                                 // 32x16 image slab from LDS
#pragma unroll
      for (int e = 0; e < 16; ++e) {
        int K = ka_index(e, h);                // slab row (tile rows -8..23)
        bv[e] = inpatch[(tr + K) * 48 + tc + hc * 16 + n];
      }
      v8f c = {};
      c = __builtin_amdgcn_wmma_f32_16x16x32_f16(false, a, false, bv,
                                                 (short)0, c, false, false);
#pragma unroll
      for (int j = 0; j < 8; ++j)
        cs[w][ridx][(j + h * 8) * 32 + hc * 16 + n] = (_Float16)c[j];
    }
  }

  // ---- row pass: S_r[m][n] = sum_{cols n-r..n+r} cs_r ----
  v8f S[4];
#pragma unroll
  for (int ridx = 0; ridx < 4; ++ridx) {
    int r = ridx + 1;
    v16h a;                                    // column-sum data as A operand
#pragma unroll
    for (int e = 0; e < 16; ++e) {
      int K = ka_index(e, h);
      a[e] = cs[w][ridx][n * 32 + K];
    }
    v16h bb;                                   // band matrix as B operand
#pragma unroll
    for (int e = 0; e < 16; ++e) {
      int K = ka_index(e, h);
      bb[e] = (K >= n + 8 - r && K <= n + 8 + r) ? (_Float16)1.f : (_Float16)0.f;
    }
    v8f c = {};
    S[ridx] = __builtin_amdgcn_wmma_f32_16x16x32_f16(false, a, false, bb,
                                                     (short)0, c, false, false);
  }

  // ---- combine: log = l1*l2 + l3 + l4, write + per-batch max ----
  float tmax = -3.4e38f;
  float* outLog = logArr + (size_t)b * IMG_PIX;
#pragma unroll
  for (int j = 0; j < 8; ++j) {
    int M = j + h * 8;
    float x = (float)inpatch[(tr + M + 8) * 48 + (tc + n + 8)];
    float l1 = (S[0][j] - 9.f * x) * (1.f / 16.f);
    float l2 = (S[1][j] - 25.f * x) * (1.f / 48.f);
    float l3 = (S[2][j] - 49.f * x) * (1.f / 96.f);
    float l4 = (S[3][j] - 81.f * x) * (1.f / 160.f);
    float lv = l1 * l2 + l3 + l4;
    outLog[(size_t)(R0 + tr + M) * IMG_N + (C0 + tc + n)] = lv;
    tmax = fmaxf(tmax, lv);
  }
  red[threadIdx.x] = tmax;
  __syncthreads();
  for (int s = 64; s > 0; s >>= 1) {
    if (threadIdx.x < s) red[threadIdx.x] = fmaxf(red[threadIdx.x], red[threadIdx.x + s]);
    __syncthreads();
  }
  if (threadIdx.x == 0) atomicMax(&logmax[b], enc_mono(red[0]));
}

// ---- K3: binarize -> 3x3 conv -> tanh -> dilate -> erode -> resize ----
__global__ __launch_bounds__(256) void k_post(const float* __restrict__ logArr,
                                              const unsigned* __restrict__ logmax,
                                              float* __restrict__ out) {
  int b = blockIdx.z;
  int r0 = blockIdx.y * 32, c0 = blockIdx.x * 32;
  float thr = dec_mono(logmax[b]) * 0.05f;
  const float* lp = logArr + (size_t)b * IMG_PIX;

  __shared__ float lg[42 * 44];   // log, origin (r0-4, c0-4)
  __shared__ float tt[40 * 44];   // tanh-threshold(segf), origin (r0-4, c0-4)
  __shared__ float dd[38 * 44];   // dilate, origin (r0-3, c0-3)
  __shared__ float ee[36 * 44];   // erode, origin (r0-2, c0-2)

  // Stage log patch: in-bounds entries stream global->LDS on the async path
  // (ASYNCcnt), halo out-of-bounds entries are zero-filled via DS stores.
  for (int idx = threadIdx.x; idx < 42 * 42; idx += 256) {
    int i = idx / 42, j = idx % 42;
    int gr = r0 - 4 + i, gc = c0 - 4 + j;
    bool inb = (gr >= 0 && gr < IMG_N && gc >= 0 && gc < IMG_N);
#if USE_ASYNC_LDS
    if (inb) {
      async_copy_b32(&lp[(size_t)gr * IMG_N + gc], &lg[i * 44 + j]);
    } else {
      lg[i * 44 + j] = 0.f;
    }
#else
    lg[i * 44 + j] = inb ? lp[(size_t)gr * IMG_N + gc] : 0.f;
#endif
  }
#if USE_ASYNC_LDS
  __builtin_amdgcn_s_wait_asynccnt(0);
#endif
  __syncthreads();

  // segf (valid 3x3 ones conv of binimg) -> threshold 2.25 -> tanh
  for (int idx = threadIdx.x; idx < 40 * 42; idx += 256) {
    int i = idx / 42, j = idx % 42;
    int gr = r0 - 4 + i, gc = c0 - 4 + j;
    float tv = -3.4e38f;                 // acts as -inf pad for dilate
    if (gr >= 0 && gr <= 1021 && gc >= 0 && gc <= 1021) {
      float s = 0.f;
#pragma unroll
      for (int di = 0; di < 3; ++di)
#pragma unroll
        for (int dj = 0; dj < 3; ++dj) {
          float v = lg[(i + di) * 44 + j + dj];
          // binimg = where(thresh(log)>0, 1, thresh(log))
          float bin = (v > thr) ? ((v > 0.f) ? 1.f : v) : 0.f;
          s += bin;
        }
      tv = (s > 2.25f) ? tanhf(s) : 0.f;
    }
    tt[i * 44 + j] = tv;
  }
  __syncthreads();

  for (int idx = threadIdx.x; idx < 38 * 42; idx += 256) {  // dilate
    int i = idx / 42, j = idx % 42;
    int gr = r0 - 3 + i, gc = c0 - 3 + j;
    float dv = 3.4e38f;                  // acts as +inf pad for erode
    if (gr >= 0 && gr <= 1021 && gc >= 0 && gc <= 1021) {
      dv = -3.4e38f;
#pragma unroll
      for (int di = 0; di < 3; ++di)
#pragma unroll
        for (int dj = 0; dj < 3; ++dj)
          dv = fmaxf(dv, tt[(i + di) * 44 + j + dj]);
    }
    dd[i * 44 + j] = dv;
  }
  __syncthreads();

  for (int idx = threadIdx.x; idx < 36 * 42; idx += 256) {  // erode
    int i = idx / 42, j = idx % 42;
    int gr = r0 - 2 + i, gc = c0 - 2 + j;
    float ev = 0.f;
    if (gr >= 0 && gr <= 1021 && gc >= 0 && gc <= 1021) {
      ev = 3.4e38f;
#pragma unroll
      for (int di = 0; di < 3; ++di)
#pragma unroll
        for (int dj = 0; dj < 3; ++dj)
          ev = fminf(ev, dd[(i + di) * 44 + j + dj]);
    }
    ee[i * 44 + j] = ev;
  }
  __syncthreads();

  // bilinear resize 1022 -> 1024, half-pixel centers, edge clamp
  const float SC = 1022.f / 1024.f;
  float* op = out + (size_t)b * IMG_PIX;
  for (int idx = threadIdx.x; idx < 32 * 32; idx += 256) {
    int i = idx / 32, j = idx % 32;
    int oy = r0 + i, ox = c0 + j;
    float sy = (oy + 0.5f) * SC - 0.5f;
    float sx = (ox + 0.5f) * SC - 0.5f;
    int iy = (int)floorf(sy), ix = (int)floorf(sx);
    float fy = sy - (float)iy, fx = sx - (float)ix;
    int iy0 = min(max(iy, 0), 1021), iy1 = min(max(iy + 1, 0), 1021);
    int ix0 = min(max(ix, 0), 1021), ix1 = min(max(ix + 1, 0), 1021);
    int ly0 = iy0 - (r0 - 2), ly1 = iy1 - (r0 - 2);
    int lx0 = ix0 - (c0 - 2), lx1 = ix1 - (c0 - 2);
    float e00 = ee[ly0 * 44 + lx0], e01 = ee[ly0 * 44 + lx1];
    float e10 = ee[ly1 * 44 + lx0], e11 = ee[ly1 * 44 + lx1];
    float v = e00 * (1.f - fy) * (1.f - fx) + e01 * (1.f - fy) * fx +
              e10 * fy * (1.f - fx) + e11 * fy * fx;
    op[(size_t)oy * IMG_N + ox] = v;
  }
}

extern "C" void kernel_launch(void* const* d_in, const int* in_sizes, int n_in,
                              void* d_out, int out_size, void* d_ws, size_t ws_size,
                              hipStream_t stream) {
  (void)in_sizes; (void)n_in; (void)out_size; (void)ws_size;
  const float* img = (const float*)d_in[0];
  float* out = (float*)d_out;
  // d_ws layout: [0..15] img-max slots, [16..31] log-max slots (u32),
  // then (256B aligned) the 64MB f32 log intermediate.
  unsigned* slots = (unsigned*)d_ws;
  float* logArr = (float*)((char*)d_ws + 256);

  k_init<<<dim3(1), dim3(32), 0, stream>>>(slots);
  k_imgmax<<<dim3(64, NBATCH), dim3(256), 0, stream>>>(img, slots);
  k_log<<<dim3(32, 32, NBATCH), dim3(128), 0, stream>>>(img, slots, slots + 16, logArr);
  k_post<<<dim3(32, 32, NBATCH), dim3(256), 0, stream>>>(logArr, slots + 16, out);
}
